// MambaLinOSS_67723044323814
// MI455X (gfx1250) — matrix operand order repaired and checked by reading, not compiled, not run on hardware
//
#include <hip/hip_runtime.h>
#include <hip/hip_bf16.h>
#include <stdint.h>

// ---------------------------------------------------------------------------
// CDNA5 (gfx1250) WMMA / TDM types
// ---------------------------------------------------------------------------
typedef __attribute__((ext_vector_type(16))) __bf16 v16bf;
typedef __attribute__((ext_vector_type(8)))  __bf16 v8bf;
typedef __attribute__((ext_vector_type(4)))  __bf16 v4bf;
typedef __attribute__((ext_vector_type(8)))  float  v8f;
typedef uint32_t v4u __attribute__((ext_vector_type(4)));
typedef int      v8i __attribute__((ext_vector_type(8)));
typedef int      v4i __attribute__((ext_vector_type(4)));

__device__ __forceinline__ __bf16 f2bf(float x) {
  uint32_t u = __builtin_bit_cast(uint32_t, x);
  uint32_t r = u + 0x7fffu + ((u >> 16) & 1u);
  uint16_t h = (uint16_t)(r >> 16);
  return __builtin_bit_cast(__bf16, h);
}

__device__ __forceinline__ v4bf f2bf4(float4 v) {
  v4bf o;
  o[0] = f2bf(v.x); o[1] = f2bf(v.y); o[2] = f2bf(v.z); o[3] = f2bf(v.w);
  return o;
}

#define TM 128
#define TN 128          // 8 WMMA tiles per wave in N
#define TK 32
#define LSTR 40         // LDS row stride (80 B): 16 dwords data + 4 dwords pad

// LDS byte offset of a __shared__ object (generic -> addrspace(3) -> u32)
__device__ __forceinline__ uint32_t lds_offset(const void* p) {
  return (uint32_t)(size_t)(__attribute__((address_space(3))) const void*)p;
}

// TDM 2-D tile load: rows x cols (cols contiguous, bf16), row stride in elems.
// LDS receives packed rows padded by 4 dwords per 16 dwords -> LSTR layout.
__device__ __forceinline__ void tdm_load_2d(uint32_t lds_off, const __bf16* gp,
                                            uint32_t tile_k, uint32_t tile_n,
                                            uint32_t stride_e,
                                            uint32_t tdim0, uint32_t tdim1) {
  uint64_t ga = (uint64_t)(size_t)gp;
  v4u g0;
  g0[0] = 1u;                                   // count=1 (valid), user mode
  g0[1] = lds_off;                              // lds_addr (bytes)
  g0[2] = (uint32_t)ga;                         // global_addr[31:0]
  g0[3] = (uint32_t)(ga >> 32) | (2u << 30);    // global_addr[56:32] | type=2
  v8i g1;
  g1[0] = (int)((1u << 16)                      // data_size = 1 (2 bytes)
              | (1u << 20)                      // pad_enable
              | (3u << 22)                      // pad_interval code 3 = 16 dwords
              | (3u << 25));                    // pad_amount  code 3 = 4 dwords
  g1[1] = (int)((tdim0 & 0xffffu) << 16);                       // tensor_dim0[15:0]
  g1[2] = (int)((tdim0 >> 16) | ((tdim1 & 0xffffu) << 16));     // td0 hi | td1 lo
  g1[3] = (int)((tdim1 >> 16) | (tile_k << 16));                // td1 hi | tile_dim0
  g1[4] = (int)(tile_n & 0xffffu);                              // tile_dim1, tile_dim2=0
  g1[5] = (int)stride_e;                                        // tensor_dim0_stride lo
  g1[6] = 0;                                                    // stride0 hi | stride1 lo
  g1[7] = 0;
  v4i z4 = {0, 0, 0, 0};
  v8i z8 = {0, 0, 0, 0, 0, 0, 0, 0};
  // clang-23 / therock-10.0 arity: (v4u, v8i, v4i, v4i, v8i, i32 cpol)
  __builtin_amdgcn_tensor_load_to_lds(g0, g1, z4, z4, z8, 0);
}

__device__ __forceinline__ v16bf frag_cat(v8bf lo, v8bf hi) {
  return __builtin_shufflevector(lo, hi, 0, 1, 2, 3, 4, 5, 6, 7,
                                         8, 9, 10, 11, 12, 13, 14, 15);
}

// A fragment (16x32 bf16): lane L -> row M=L%16; two aligned 16-B chunks.
__device__ __forceinline__ v16bf load_frag_a(const __bf16* As, int lane) {
  const int r = lane & 15, hi = lane >> 4;
  const __bf16* row = As + r * LSTR;
  v8bf lo = *(const v8bf*)(row + hi * 8);
  v8bf hh = *(const v8bf*)(row + 16 + hi * 8);
  return frag_cat(lo, hh);
}

// B fragment (32x16 bf16, KxN) from a TRANSPOSED tile Bt[n][k]:
// lane L -> col N=L%16; two aligned 16-B chunks (same k order as A).
__device__ __forceinline__ v16bf load_frag_b(const __bf16* Bt, int lane) {
  const int c = lane & 15, hi = lane >> 4;
  const __bf16* row = Bt + c * LSTR;
  v8bf lo = *(const v8bf*)(row + hi * 8);
  v8bf hh = *(const v8bf*)(row + 16 + hi * 8);
  return frag_cat(lo, hh);
}

// ---------------------------------------------------------------------------
// Pre-pass: f32 -> bf16 conversions (vectorized), optional scale
// ---------------------------------------------------------------------------
__global__ __launch_bounds__(256)
void cvt1d_kernel(const float* __restrict__ in, __bf16* __restrict__ out,
                  long n4, float scale)
{
  long i = (long)blockIdx.x * blockDim.x + threadIdx.x;
  long stride = (long)gridDim.x * blockDim.x;
  for (; i < n4; i += stride) {
    float4 v = ((const float4*)in)[i];
    v.x *= scale; v.y *= scale; v.z *= scale; v.w *= scale;
    ((v4bf*)out)[i] = f2bf4(v);
  }
}

// (K,N) f32 -> (N,K) bf16 (used to re-lay W_dt as N-major)
__global__ __launch_bounds__(256)
void tcvt_kernel(const float* __restrict__ in, __bf16* __restrict__ out, int K, int N)
{
  int t = blockIdx.x * 256 + threadIdx.x;
  if (t < K * N) {
    int nn = t / K, kk = t - nn * K;
    out[(size_t)nn * K + kk] = f2bf(in[(size_t)kk * N + nn]);
  }
}

// ---------------------------------------------------------------------------
// Phase 1 (fused): [dt | Bu_re | Bu_im] = u_bf @ Wcat^T, Wcat is (3P, H) bf16
// A tiles and B tiles both staged by the Tensor Data Mover, double-buffered.
// ---------------------------------------------------------------------------
__global__ __launch_bounds__(256)
void gemm_in_kernel(const __bf16* __restrict__ uB, const __bf16* __restrict__ Wcat,
                    float* __restrict__ dt, float* __restrict__ buRe,
                    float* __restrict__ buIm, int BL, int H, int P)
{
  __shared__ alignas(16) __bf16 As[2][TM][LSTR];
  __shared__ alignas(16) __bf16 Bs[2][TN][LSTR];

  const int tid  = threadIdx.x;
  const int lane = tid & 31;
  const int wave = tid >> 5;
  const int m0 = blockIdx.x * TM;
  const int n0 = blockIdx.y * TN;     // index in concatenated 3P space

  v8f vz = {0.f, 0.f, 0.f, 0.f, 0.f, 0.f, 0.f, 0.f};
  v8f acc[8] = {vz, vz, vz, vz, vz, vz, vz, vz};

  const int nk = H / TK;
  if (wave == 0) {  // prologue: DMA tile 0 into buffer 0
    tdm_load_2d(lds_offset(&As[0][0][0]), uB + (size_t)m0 * H, TK, TM, H, H, TM);
    tdm_load_2d(lds_offset(&Bs[0][0][0]), Wcat + (size_t)n0 * H, TK, TN, H, H, TN);
  }

  for (int kb = 0; kb < nk; ++kb) {
    const int cur = kb & 1;
    if (wave == 0) __builtin_amdgcn_s_wait_tensorcnt(0);
    __syncthreads();                 // buffer `cur` ready; prior compute done
    if (wave == 0 && kb + 1 < nk) {  // DMA next tile into the other buffer
      const int k1 = (kb + 1) * TK;
      tdm_load_2d(lds_offset(&As[cur ^ 1][0][0]), uB + (size_t)m0 * H + k1,
                  TK, TM, H, H, TM);
      tdm_load_2d(lds_offset(&Bs[cur ^ 1][0][0]), Wcat + (size_t)n0 * H + k1,
                  TK, TN, H, H, TN);
    }
    v16bf a = load_frag_a(&As[cur][wave * 16][0], lane);
#pragma unroll
    for (int j = 0; j < 8; ++j) {
      v16bf b = load_frag_b(&Bs[cur][j * 16][0], lane);
      acc[j] = __builtin_amdgcn_wmma_f32_16x16x32_bf16(
          false, a, false, b, (short)0, acc[j], false, false);
    }
  }

  // epilogue: route by head (0:dt+sigmoid, 1:buRe, 2:buIm)
  const int head  = n0 / P;               // uniform per block (TN divides P)
  float* outp = head == 0 ? dt : (head == 1 ? buRe : buIm);
  const int ncol0 = n0 - head * P;
  const int hi2   = lane >> 4;
  const int nloc  = lane & 15;
#pragma unroll
  for (int j = 0; j < 8; ++j) {
    const int col = ncol0 + j * 16 + nloc;
#pragma unroll
    for (int i = 0; i < 8; ++i) {
      const int m = m0 + wave * 16 + hi2 * 8 + i;
      float v = acc[j][i];
      if (head == 0) v = 1.0f / (1.0f + __expf(-v));
      outp[(size_t)m * P + col] = v;
    }
  }
}

// ---------------------------------------------------------------------------
// Phase 2: chunked linear-recurrence scan  s_l = M_l s_{l-1} + F_l
// ---------------------------------------------------------------------------
__device__ __forceinline__ void make_MF(float d, float Ad, float br, float bi,
                                        float& M11, float& M12, float& M21, float& M22,
                                        float& f1r, float& f1i, float& f2r, float& f2i)
{
  float dd = d * d * Ad;
  float S  = 1.0f / (1.0f + dd);
  M11 = 1.0f - dd * S;
  M12 = -d * Ad * S;
  M21 = d * S;
  M22 = S;
  float db_r = d * br, db_i = d * bi;
  f1r = M11 * db_r; f1i = M11 * db_i;
  f2r = M21 * db_r; f2i = M21 * db_i;
}

__global__ __launch_bounds__(256)
void scan_chunk_kernel(const float* __restrict__ dt, const float* __restrict__ buRe,
                       const float* __restrict__ buIm, const float* __restrict__ A_diag,
                       float* __restrict__ chA, float* __restrict__ chB,
                       int L, int P, int Lc)
{
  const int b = blockIdx.x, c = blockIdx.y, p = threadIdx.x;
  const float Ad = fmaxf(A_diag[p], 0.0f);
  float a11 = 1.f, a12 = 0.f, a21 = 0.f, a22 = 1.f;
  float b1r = 0.f, b1i = 0.f, b2r = 0.f, b2i = 0.f;

  size_t base = ((size_t)b * L + (size_t)c * Lc) * P + p;
  for (int l = 0; l < Lc; ++l, base += P) {
    float M11, M12, M21, M22, f1r, f1i, f2r, f2i;
    make_MF(dt[base], Ad, buRe[base], buIm[base], M11, M12, M21, M22, f1r, f1i, f2r, f2i);
    float na11 = M11 * a11 + M12 * a21, na12 = M11 * a12 + M12 * a22;
    float na21 = M21 * a11 + M22 * a21, na22 = M21 * a12 + M22 * a22;
    a11 = na11; a12 = na12; a21 = na21; a22 = na22;
    float n1r = M11 * b1r + M12 * b2r + f1r, n1i = M11 * b1i + M12 * b2i + f1i;
    float n2r = M21 * b1r + M22 * b2r + f2r, n2i = M21 * b1i + M22 * b2i + f2i;
    b1r = n1r; b1i = n1i; b2r = n2r; b2i = n2i;
  }
  size_t s = (((size_t)b * gridDim.y + c) * P + p) * 4;
  chA[s + 0] = a11; chA[s + 1] = a12; chA[s + 2] = a21; chA[s + 3] = a22;
  chB[s + 0] = b1r; chB[s + 1] = b1i; chB[s + 2] = b2r; chB[s + 3] = b2i;
}

__global__ __launch_bounds__(256)
void scan_prefix_kernel(const float* __restrict__ chA, const float* __restrict__ chB,
                        float* __restrict__ entry, int NC, int P)
{
  const int b = blockIdx.x, p = threadIdx.x;
  float e1r = 0.f, e1i = 0.f, e2r = 0.f, e2i = 0.f;
  for (int c = 0; c < NC; ++c) {
    size_t s = (((size_t)b * NC + c) * P + p) * 4;
    entry[s + 0] = e1r; entry[s + 1] = e1i; entry[s + 2] = e2r; entry[s + 3] = e2i;
    float a11 = chA[s + 0], a12 = chA[s + 1], a21 = chA[s + 2], a22 = chA[s + 3];
    float c1r = chB[s + 0], c1i = chB[s + 1], c2r = chB[s + 2], c2i = chB[s + 3];
    float n1r = a11 * e1r + a12 * e2r + c1r, n1i = a11 * e1i + a12 * e2i + c1i;
    float n2r = a21 * e1r + a22 * e2r + c2r, n2i = a21 * e1i + a22 * e2i + c2i;
    e1r = n1r; e1i = n1i; e2r = n2r; e2i = n2i;
  }
}

// Re-applies recurrence from corrected entry state; writes x2 in place
// (re -> dtBuf, im -> burBuf).
__global__ __launch_bounds__(256)
void scan_apply_kernel(float* dtBuf, float* burBuf, const float* __restrict__ buiBuf,
                       const float* __restrict__ A_diag, const float* __restrict__ entry,
                       int L, int P, int Lc)
{
  const int b = blockIdx.x, c = blockIdx.y, p = threadIdx.x;
  const int NC = gridDim.y;
  const float Ad = fmaxf(A_diag[p], 0.0f);
  size_t s = (((size_t)b * NC + c) * P + p) * 4;
  float b1r = entry[s + 0], b1i = entry[s + 1], b2r = entry[s + 2], b2i = entry[s + 3];

  size_t base = ((size_t)b * L + (size_t)c * Lc) * P + p;
  for (int l = 0; l < Lc; ++l, base += P) {
    float d = dtBuf[base], br = burBuf[base], bi = buiBuf[base];
    float M11, M12, M21, M22, f1r, f1i, f2r, f2i;
    make_MF(d, Ad, br, bi, M11, M12, M21, M22, f1r, f1i, f2r, f2i);
    float n1r = M11 * b1r + M12 * b2r + f1r, n1i = M11 * b1i + M12 * b2i + f1i;
    float n2r = M21 * b1r + M22 * b2r + f2r, n2i = M21 * b1i + M22 * b2i + f2i;
    b1r = n1r; b1i = n1i; b2r = n2r; b2i = n2i;
    dtBuf[base]  = b2r;   // x_re
    burBuf[base] = b2i;   // x_im
  }
}

// ---------------------------------------------------------------------------
// Phase 3: y[m,h] = sum_p xRe*C_re - xIm*C_im + D[h]*u[m,h]
// A (x, f32) staged manually with conversion; B (bf16 C tiles) via TDM.
// ---------------------------------------------------------------------------
__global__ __launch_bounds__(256)
void gemm_out_kernel(const float* __restrict__ xRe, const float* __restrict__ xIm,
                     const __bf16* __restrict__ CreB, const __bf16* __restrict__ CimB,
                     const float* __restrict__ Dvec, const float* __restrict__ u,
                     float* __restrict__ y, int BL, int H, int P)
{
  __shared__ alignas(16) __bf16 Ar[TM][LSTR];
  __shared__ alignas(16) __bf16 Ai[TM][LSTR];
  __shared__ alignas(16) __bf16 Brs[2][TN][LSTR];
  __shared__ alignas(16) __bf16 Bis[2][TN][LSTR];   // holds -C_im (pre-negated)

  const int tid  = threadIdx.x;
  const int lane = tid & 31;
  const int wave = tid >> 5;
  const int m0 = blockIdx.x * TM;
  const int n0 = blockIdx.y * TN;     // n indexes H

  v8f vz = {0.f, 0.f, 0.f, 0.f, 0.f, 0.f, 0.f, 0.f};
  v8f acc[8] = {vz, vz, vz, vz, vz, vz, vz, vz};

  const int nk = P / TK;
  if (wave == 0) {
    tdm_load_2d(lds_offset(&Brs[0][0][0]), CreB + (size_t)n0 * P, TK, TN, P, P, TN);
    tdm_load_2d(lds_offset(&Bis[0][0][0]), CimB + (size_t)n0 * P, TK, TN, P, P, TN);
  }

  for (int kb = 0; kb < nk; ++kb) {
    const int cur = kb & 1;
    const int k0 = kb * TK;
    __syncthreads();                 // prior compute done before A overwrite
    for (int t = tid; t < (TM * TK) / 4; t += 256) {
      int r = t >> 3, k = (t & 7) * 4;
      size_t gi = (size_t)(m0 + r) * P + (k0 + k);
      *(v4bf*)&Ar[r][k] = f2bf4(*(const float4*)&xRe[gi]);
      *(v4bf*)&Ai[r][k] = f2bf4(*(const float4*)&xIm[gi]);
    }
    if (wave == 0) __builtin_amdgcn_s_wait_tensorcnt(0);
    __syncthreads();                 // A staged + B tile `cur` visible
    if (wave == 0 && kb + 1 < nk) {
      const int k1 = (kb + 1) * TK;
      tdm_load_2d(lds_offset(&Brs[cur ^ 1][0][0]), CreB + (size_t)n0 * P + k1,
                  TK, TN, P, P, TN);
      tdm_load_2d(lds_offset(&Bis[cur ^ 1][0][0]), CimB + (size_t)n0 * P + k1,
                  TK, TN, P, P, TN);
    }
    v16bf ar = load_frag_a(&Ar[wave * 16][0], lane);
    v16bf ai = load_frag_a(&Ai[wave * 16][0], lane);
#pragma unroll
    for (int j = 0; j < 8; ++j) {
      v16bf br = load_frag_b(&Brs[cur][j * 16][0], lane);
      acc[j] = __builtin_amdgcn_wmma_f32_16x16x32_bf16(
          false, ar, false, br, (short)0, acc[j], false, false);
      v16bf bi = load_frag_b(&Bis[cur][j * 16][0], lane);
      acc[j] = __builtin_amdgcn_wmma_f32_16x16x32_bf16(
          false, ai, false, bi, (short)0, acc[j], false, false);
    }
  }

  const int hi2  = lane >> 4;
  const int nloc = lane & 15;
#pragma unroll
  for (int j = 0; j < 8; ++j) {
    const int n = n0 + j * 16 + nloc;
    const float dn = Dvec[n];
#pragma unroll
    for (int i = 0; i < 8; ++i) {
      const int m = m0 + wave * 16 + hi2 * 8 + i;
      y[(size_t)m * H + n] = acc[j][i] + dn * u[(size_t)m * H + n];
    }
  }
}

// ---------------------------------------------------------------------------
// Host-side launch
// ---------------------------------------------------------------------------
static inline int cvt_grid(long n4) {
  long g = (n4 + 255) / 256;
  return (int)(g > 4096 ? 4096 : g);
}

extern "C" void kernel_launch(void* const* d_in, const int* in_sizes, int n_in,
                              void* d_out, int out_size, void* d_ws, size_t ws_size,
                              hipStream_t stream) {
  const float* u    = (const float*)d_in[0];
  const float* A_d  = (const float*)d_in[1];
  const float* B_re = (const float*)d_in[2];
  const float* B_im = (const float*)d_in[3];
  const float* C_re = (const float*)d_in[4];
  const float* C_im = (const float*)d_in[5];
  const float* Dv   = (const float*)d_in[6];
  const float* W_dt = (const float*)d_in[7];
  float* y = (float*)d_out;

  const int P  = in_sizes[1];          // 256
  const int H  = in_sizes[6];          // 768
  const int BL = in_sizes[0] / H;      // 32768
  const int B  = 8;                    // per reference setup
  const int L  = BL / B;               // 4096
  const int NC = 64;                   // scan chunks
  const int Lc = L / NC;               // 64

  float* ws = (float*)d_ws;
  const size_t n = (size_t)BL * P;
  float* dt    = ws;                   // aliased as x_re after scan
  float* buRe  = ws + n;               // aliased as x_im after scan
  float* buIm  = ws + 2 * n;
  float* chA   = ws + 3 * n;
  const size_t nch = (size_t)B * NC * P * 4;
  float* chB   = chA + nch;
  float* entry = chB + nch;
  __bf16* u_bf = (__bf16*)(entry + nch);
  __bf16* Wcat = u_bf + (size_t)BL * H;       // (3P, H): [W_dt^T | B_re | B_im]
  __bf16* CreB = Wcat + (size_t)3 * P * H;
  __bf16* CimB = CreB + (size_t)H * P;        // stores -C_im

  dim3 blk(256);

  // --- pre-pass: one-time bf16 conversions -------------------------------
  const long n4u = (long)BL * H / 4;
  const long n4w = (long)P * H / 4;
  cvt1d_kernel<<<cvt_grid(n4u), blk, 0, stream>>>(u,    u_bf,              n4u,  1.f);
  tcvt_kernel <<<(H * P + 255) / 256, blk, 0, stream>>>(W_dt, Wcat, H, P);  // (H,P)->(P,H)
  cvt1d_kernel<<<cvt_grid(n4w), blk, 0, stream>>>(B_re, Wcat + (size_t)P * H,     n4w,  1.f);
  cvt1d_kernel<<<cvt_grid(n4w), blk, 0, stream>>>(B_im, Wcat + (size_t)2 * P * H, n4w,  1.f);
  cvt1d_kernel<<<cvt_grid(n4w), blk, 0, stream>>>(C_re, CreB,              n4w,  1.f);
  cvt1d_kernel<<<cvt_grid(n4w), blk, 0, stream>>>(C_im, CimB,              n4w, -1.f);

  // --- phase 1: fused [dt | Bu_re | Bu_im] GEMM (TDM-staged) -------------
  dim3 g1(BL / TM, (3 * P) / TN);
  gemm_in_kernel<<<g1, blk, 0, stream>>>(u_bf, Wcat, dt, buRe, buIm, BL, H, P);

  // --- phase 2: chunked scan ---------------------------------------------
  scan_chunk_kernel <<<dim3(B, NC), dim3(P), 0, stream>>>(dt, buRe, buIm, A_d, chA, chB, L, P, Lc);
  scan_prefix_kernel<<<dim3(B),     dim3(P), 0, stream>>>(chA, chB, entry, NC, P);
  scan_apply_kernel <<<dim3(B, NC), dim3(P), 0, stream>>>(dt, buRe, buIm, A_d, entry, L, P, Lc);

  // --- phase 3: output GEMM (B tiles TDM-staged) -------------------------
  dim3 g3(BL / TM, H / TN);
  gemm_out_kernel<<<g3, blk, 0, stream>>>(dt /*x_re*/, buRe /*x_im*/, CreB, CimB,
                                          Dv, u, y, BL, H, P);
}
